// Attention_37598143710100
// MI455X (gfx1250) — matrix operand order repaired and checked by reading, not compiled, not run on hardware
//
#include <hip/hip_runtime.h>
#include <cstdint>

typedef __attribute__((ext_vector_type(16))) __bf16 v16bf;
typedef __attribute__((ext_vector_type(8)))  float  v8f;

#define B_ 4
#define L_ 4096
#define C_ 1024
#define H_ 16
#define D_ 64
#define M_TOT (B_ * L_)   /* 16384 */
#define N_QKV (3 * C_)    /* 3072  */

// 16-bit A-matrix 16x32 fragment in this code's contiguous k order (k = 16*half + e, not the ISA table order). lane half = lane>>4, m = lane&15.
// element e of the v16bf lives in VGPR e/2 (lo/hi by e&1); K mapping below.
__device__ __forceinline__ int a_frag_k(int e, int half) {
  const int r = e >> 1, p = e & 1;
  return half * 16 + 2 * r + p;
}

__device__ __forceinline__ v8f wmma_bf16(v16bf a, v16bf b, v8f c) {
  return __builtin_amdgcn_wmma_f32_16x16x32_bf16(false, a, false, b, (short)0, c,
                                                 false, false);
}

// LDS byte offset of a __shared__ object (addrspace(3) pointers are 32-bit
// offsets from the workgroup's LDS base, which the HW adds for async ops).
__device__ __forceinline__ unsigned lds_off(const void* p) {
  return (unsigned)(size_t)(__attribute__((address_space(3))) const void*)p;
}

// CDNA5 async Global->LDS copy (ASYNCcnt-tracked, no VGPR round trip).
// GV mode: VDST = LDS addr VGPR, VADDR = 64-bit global address.
__device__ __forceinline__ void async_copy_b128(unsigned lds_dst,
                                                const void* gsrc) {
  asm volatile("global_load_async_to_lds_b128 %0, %1, off"
               :: "v"(lds_dst), "v"(gsrc)
               : "memory");
}

__device__ __forceinline__ void wait_async0() {
  asm volatile("s_wait_asynccnt 0x0" ::: "memory");
}

// ---------------------------------------------------------------------------
// K1: qkv = x @ W_qkv^T, scatter into bf16 q/k/v [B,H,L,D].
// Block = 64x64 tile, 4 waves (2x2), wave = 32x32 (2x2 WMMA 16x16), K step 32.
// (f32 sources need VALU conversion, so staging stays on the load+cvt path.)
// ---------------------------------------------------------------------------
__global__ __launch_bounds__(128) void qkv_gemm_kernel(
    const float* __restrict__ x, const float* __restrict__ Wqkv,
    __bf16* __restrict__ qb, __bf16* __restrict__ kb, __bf16* __restrict__ vb) {
  __shared__ __bf16 As[64][40];
  __shared__ __bf16 Bs[64][40];
  const int m0 = blockIdx.x * 64, n0 = blockIdx.y * 64;
  const int tid = threadIdx.x;
  const int lane = tid & 31, wave = tid >> 5;
  const int half = lane >> 4, lm = lane & 15;
  const int wm = (wave >> 1) * 32, wn = (wave & 1) * 32;

  v8f acc[2][2];
  acc[0][0] = {}; acc[0][1] = {}; acc[1][0] = {}; acc[1][1] = {};

  const int arow = tid >> 1, acol = (tid & 1) * 16;

  for (int k0 = 0; k0 < C_; k0 += 32) {
    __syncthreads();
    const float* ag = x + (size_t)(m0 + arow) * C_ + k0 + acol;
    const float* bg = Wqkv + (size_t)(n0 + arow) * C_ + k0 + acol;
#pragma unroll
    for (int j = 0; j < 16; ++j) {
      As[arow][acol + j] = (__bf16)ag[j];
      Bs[arow][acol + j] = (__bf16)bg[j];
    }
    if (k0 + 32 < C_) {  // global_prefetch_b8 for next K tile
      __builtin_prefetch(ag + 32, 0, 0);
      __builtin_prefetch(bg + 32, 0, 0);
    }
    __syncthreads();

    v16bf a0, a1, b0, b1;
#pragma unroll
    for (int e = 0; e < 16; ++e) {
      const int ka = a_frag_k(e, half);
      const int kk = half * 16 + e;  // B-fragment K index (K x N, N = lm)
      a0[e] = As[wm + lm][ka];
      a1[e] = As[wm + 16 + lm][ka];
      b0[e] = Bs[wn + lm][kk];
      b1[e] = Bs[wn + 16 + lm][kk];
    }
    acc[0][0] = wmma_bf16(a0, b0, acc[0][0]);
    acc[0][1] = wmma_bf16(a0, b1, acc[0][1]);
    acc[1][0] = wmma_bf16(a1, b0, acc[1][0]);
    acc[1][1] = wmma_bf16(a1, b1, acc[1][1]);
  }

  // C layout: VGPR r -> row r + 8*half, col = lm. Scatter to (part,h) head.
#pragma unroll
  for (int i = 0; i < 2; ++i)
#pragma unroll
    for (int j = 0; j < 2; ++j)
#pragma unroll
      for (int r = 0; r < 8; ++r) {
        const int m = m0 + wm + i * 16 + r + 8 * half;
        const int e = n0 + wn + j * 16 + lm;
        const int part = e >> 10, rem = e & 1023;
        const int h = rem >> 6, d = rem & 63;
        const int b = m >> 12, l = m & (L_ - 1);
        __bf16* dst = (part == 0) ? qb : (part == 1) ? kb : vb;
        dst[(((size_t)(b * H_ + h)) * L_ + l) * D_ + d] = (__bf16)acc[i][j][r];
      }
}

// ---------------------------------------------------------------------------
// K2: in-place RMS-norm + RoPE on one [D] row per thread (q or k).
// ---------------------------------------------------------------------------
__global__ __launch_bounds__(256) void norm_rope_kernel(
    __bf16* __restrict__ t, const float* __restrict__ scale,
    const float* __restrict__ cosT, const float* __restrict__ sinT) {
  const int idx = blockIdx.x * blockDim.x + threadIdx.x;  // (b*H+h)*L + l
  if (idx >= B_ * H_ * L_) return;
  const int l = idx & (L_ - 1);
  __bf16* row = t + (size_t)idx * D_;
  float v[D_];
  float ss = 0.f;
#pragma unroll
  for (int d = 0; d < D_; ++d) {
    v[d] = (float)row[d];
    ss += v[d] * v[d];
  }
  const float inv = rsqrtf(ss * (1.f / D_) + 1e-6f);
  const float* cr = cosT + (size_t)l * D_;
  const float* sr = sinT + (size_t)l * D_;
#pragma unroll
  for (int d = 0; d < 32; ++d) {
    const float a  = v[d]      * inv * scale[d];
    const float b2 = v[d + 32] * inv * scale[d + 32];
    row[d]      = (__bf16)(a * cr[d] - b2 * sr[d]);        // rotate_half: -t2
    row[d + 32] = (__bf16)(b2 * cr[d + 32] + a * sr[d + 32]);  // +t1
  }
}

// ---------------------------------------------------------------------------
// K3: flash attention. Block = (b,h, 64 q-rows), 4 waves; wave owns 16 q-rows
// with a 16x64 fp32 output accumulator (4 WMMA C tiles). Key tiles of 32.
// K/V tiles staged with CDNA5 async Global->LDS (ASYNCcnt) copies.
// ---------------------------------------------------------------------------
__global__ __launch_bounds__(128) void flash_attn_kernel(
    const __bf16* __restrict__ qb, const __bf16* __restrict__ kb,
    const __bf16* __restrict__ vb, __bf16* __restrict__ ob) {
  __shared__ __bf16 Ks[32][72];
  __shared__ __bf16 Vs[32][72];
  __shared__ __bf16 Ps[4][16][40];
  const int bh = blockIdx.y;
  const int q0 = blockIdx.x * 64;
  const int tid = threadIdx.x;
  const int lane = tid & 31, wave = tid >> 5;
  const int half = lane >> 4, lm = lane & 15;
  const size_t base = (size_t)bh * L_ * D_;

  // Q fragments (A-matrix, K dim = D = 64 -> two 32-chunks), loaded once.
  v16bf qf[2];
  {
    const int qrow = q0 + wave * 16 + lm;
#pragma unroll
    for (int c = 0; c < 2; ++c)
#pragma unroll
      for (int e = 0; e < 16; ++e)
        qf[c][e] = qb[base + (size_t)qrow * D_ + c * 32 + a_frag_k(e, half)];
  }

  v8f o[4];
  o[0] = {}; o[1] = {}; o[2] = {}; o[3] = {};
  float mrow[8], lrow[8];
#pragma unroll
  for (int r = 0; r < 8; ++r) { mrow[r] = -1e30f; lrow[r] = 0.f; }
  const float sscale = 0.125f;  // D^-0.5

  const int ldrow = tid >> 2, lddb = (tid & 3) * 16;
  const unsigned kdst0 = lds_off(&Ks[ldrow][lddb]);
  const unsigned kdst1 = lds_off(&Ks[ldrow][lddb + 8]);
  const unsigned vdst0 = lds_off(&Vs[ldrow][lddb]);
  const unsigned vdst1 = lds_off(&Vs[ldrow][lddb + 8]);

  for (int kt = 0; kt < L_; kt += 32) {
    __syncthreads();  // previous tile fully consumed before overwrite
    {
      const __bf16* srck = kb + base + (size_t)(kt + ldrow) * D_ + lddb;
      const __bf16* srcv = vb + base + (size_t)(kt + ldrow) * D_ + lddb;
      async_copy_b128(kdst0, srck);
      async_copy_b128(kdst1, srck + 8);
      async_copy_b128(vdst0, srcv);
      async_copy_b128(vdst1, srcv + 8);
    }
    wait_async0();    // own wave's copies complete
    __syncthreads();  // all waves' copies visible

    // S = Q K^T : two 16x16 C tiles (keys 0-15, 16-31)
    v8f s0 = {}, s1 = {};
#pragma unroll
    for (int c = 0; c < 2; ++c) {
      v16bf k0f, k1f;
#pragma unroll
      for (int e = 0; e < 16; ++e) {
        const int kk = c * 32 + half * 16 + e;  // K-dim = d
        k0f[e] = Ks[lm][kk];
        k1f[e] = Ks[16 + lm][kk];
      }
      s0 = wmma_bf16(qf[c], k0f, s0);
      s1 = wmma_bf16(qf[c], k1f, s1);
    }

    // Online softmax; row r+8*half lives across the 16 lanes of this half.
#pragma unroll
    for (int r = 0; r < 8; ++r) {
      const float a = s0[r] * sscale, b2 = s1[r] * sscale;
      float mx = fmaxf(a, b2);
#pragma unroll
      for (int off = 1; off < 16; off <<= 1)
        mx = fmaxf(mx, __shfl_xor(mx, off, 32));
      const float mnew = fmaxf(mrow[r], mx);
      const float alpha = __expf(mrow[r] - mnew);
      const float e0 = __expf(a - mnew), e1 = __expf(b2 - mnew);
      float ssum = e0 + e1;
#pragma unroll
      for (int off = 1; off < 16; off <<= 1)
        ssum += __shfl_xor(ssum, off, 32);
      lrow[r] = lrow[r] * alpha + ssum;
      mrow[r] = mnew;
#pragma unroll
      for (int j = 0; j < 4; ++j) o[j][r] *= alpha;
      const int pm = r + 8 * half;  // C-layout row -> LDS, re-read as A frag
      Ps[wave][pm][lm]      = (__bf16)e0;
      Ps[wave][pm][16 + lm] = (__bf16)e1;
    }
    __syncthreads();

    // O += P (16x32) x V (32x64)
    v16bf pf;
#pragma unroll
    for (int e = 0; e < 16; ++e) pf[e] = Ps[wave][lm][a_frag_k(e, half)];
#pragma unroll
    for (int j = 0; j < 4; ++j) {
      v16bf vf;
#pragma unroll
      for (int e = 0; e < 16; ++e) vf[e] = Vs[half * 16 + e][j * 16 + lm];
      o[j] = wmma_bf16(pf, vf, o[j]);
    }
  }

  const int b = bh >> 4, h = bh & (H_ - 1);
#pragma unroll
  for (int r = 0; r < 8; ++r) {
    const float inv = 1.f / lrow[r];
    const size_t row = (size_t)b * L_ + q0 + wave * 16 + r + 8 * half;
#pragma unroll
    for (int j = 0; j < 4; ++j)
      ob[row * C_ + h * D_ + j * 16 + lm] = (__bf16)(o[j][r] * inv);
  }
}

// ---------------------------------------------------------------------------
// K4: out = att @ W_proj^T + b_proj (fp32 out). Same tiling as K1; the
// already-bf16 A tile is staged with async Global->LDS copies.
// ---------------------------------------------------------------------------
__global__ __launch_bounds__(128) void proj_gemm_kernel(
    const __bf16* __restrict__ att, const float* __restrict__ Wp,
    const float* __restrict__ bias, float* __restrict__ out) {
  __shared__ __bf16 As[64][40];
  __shared__ __bf16 Bs[64][40];
  const int m0 = blockIdx.x * 64, n0 = blockIdx.y * 64;
  const int tid = threadIdx.x;
  const int lane = tid & 31, wave = tid >> 5;
  const int half = lane >> 4, lm = lane & 15;
  const int wm = (wave >> 1) * 32, wn = (wave & 1) * 32;

  v8f acc[2][2];
  acc[0][0] = {}; acc[0][1] = {}; acc[1][0] = {}; acc[1][1] = {};

  const int arow = tid >> 1, acol = (tid & 1) * 16;
  const unsigned adst0 = lds_off(&As[arow][acol]);
  const unsigned adst1 = lds_off(&As[arow][acol + 8]);

  for (int k0 = 0; k0 < C_; k0 += 32) {
    __syncthreads();
    {
      const __bf16* ag = att + (size_t)(m0 + arow) * C_ + k0 + acol;
      async_copy_b128(adst0, ag);       // bf16 tile: pure async copy
      async_copy_b128(adst1, ag + 8);
      const float* bg = Wp + (size_t)(n0 + arow) * C_ + k0 + acol;
#pragma unroll
      for (int j = 0; j < 16; ++j) Bs[arow][acol + j] = (__bf16)bg[j];
      if (k0 + 32 < C_) __builtin_prefetch(bg + 32, 0, 0);
    }
    wait_async0();
    __syncthreads();

    v16bf a0, a1, b0, b1;
#pragma unroll
    for (int e = 0; e < 16; ++e) {
      const int ka = a_frag_k(e, half);
      const int kk = half * 16 + e;
      a0[e] = As[wm + lm][ka];
      a1[e] = As[wm + 16 + lm][ka];
      b0[e] = Bs[wn + lm][kk];
      b1[e] = Bs[wn + 16 + lm][kk];
    }
    acc[0][0] = wmma_bf16(a0, b0, acc[0][0]);
    acc[0][1] = wmma_bf16(a0, b1, acc[0][1]);
    acc[1][0] = wmma_bf16(a1, b0, acc[1][0]);
    acc[1][1] = wmma_bf16(a1, b1, acc[1][1]);
  }

#pragma unroll
  for (int i = 0; i < 2; ++i)
#pragma unroll
    for (int j = 0; j < 2; ++j)
#pragma unroll
      for (int r = 0; r < 8; ++r) {
        const int m = m0 + wm + i * 16 + r + 8 * half;
        const int n = n0 + wn + j * 16 + lm;
        out[(size_t)m * C_ + n] = acc[i][j][r] + bias[n];
      }
}

// ---------------------------------------------------------------------------
extern "C" void kernel_launch(void* const* d_in, const int* in_sizes, int n_in,
                              void* d_out, int out_size, void* d_ws,
                              size_t ws_size, hipStream_t stream) {
  const float* x       = (const float*)d_in[0];
  const float* W_qkv   = (const float*)d_in[1];
  const float* q_scale = (const float*)d_in[2];
  const float* k_scale = (const float*)d_in[3];
  const float* W_proj  = (const float*)d_in[4];
  const float* b_proj  = (const float*)d_in[5];
  const float* cosT    = (const float*)d_in[6];
  const float* sinT    = (const float*)d_in[7];
  float* out = (float*)d_out;

  const size_t headElems = (size_t)B_ * H_ * L_ * D_;  // 16M elems
  char* ws = (char*)d_ws;                               // 128 MB used
  __bf16* qb  = (__bf16*)(ws);
  __bf16* kb  = (__bf16*)(ws + headElems * 2);
  __bf16* vb  = (__bf16*)(ws + headElems * 4);
  __bf16* att = (__bf16*)(ws + headElems * 6);

  qkv_gemm_kernel<<<dim3(M_TOT / 64, N_QKV / 64), 128, 0, stream>>>(
      x, W_qkv, qb, kb, vb);
  const int rows = B_ * H_ * L_;
  norm_rope_kernel<<<(rows + 255) / 256, 256, 0, stream>>>(qb, q_scale, cosT, sinT);
  norm_rope_kernel<<<(rows + 255) / 256, 256, 0, stream>>>(kb, k_scale, cosT, sinT);
  flash_attn_kernel<<<dim3(L_ / 64, B_ * H_), 128, 0, stream>>>(qb, kb, vb, att);
  proj_gemm_kernel<<<dim3(M_TOT / 64, C_ / 64), 128, 0, stream>>>(
      att, W_proj, b_proj, out);
}
